// SelfAttention2D_Block_N_69501160784168
// MI455X (gfx1250) — matrix operand order repaired and checked by reading, not compiled, not run on hardware
//
#include <hip/hip_runtime.h>
#include <cstdint>

// ---------------- problem constants ----------------
#define Nn   256
#define Tt   12
#define Ff   32
#define TD   8
#define Dd   64
#define KTD  6144   // T_dim * T * D   (k,t,d) flattened contraction dim
#define NTD  2048   // N * T_dim

typedef __attribute__((ext_vector_type(16))) __bf16 v16bf;
typedef __attribute__((ext_vector_type(8)))  __bf16 v8bf;
typedef __attribute__((ext_vector_type(8)))  float  v8f;
typedef __attribute__((ext_vector_type(4)))  unsigned int u32x4;
typedef __attribute__((ext_vector_type(8)))  int    i32x8;
typedef __attribute__((ext_vector_type(4)))  int    i32x4;

#if __has_builtin(__builtin_amdgcn_tensor_load_to_lds) && __has_builtin(__builtin_amdgcn_s_wait_tensorcnt)
#define HAVE_TDM 1
#else
#define HAVE_TDM 0
#endif

// ---------------- workspace layout (bytes, all 256B aligned) ----------------
#define WS_Q    0                         // 256 x 6144 bf16 : Q[i][(k,t,d)]
#define WS_K    (WS_Q   + Nn*KTD*2)       // 256 x 6144 bf16 : K[j][(k,s,d)]
#define WS_VT   (WS_K   + Nn*KTD*2)       // 6144 x 256 bf16 : V^T[(k,t,d)][j]
#define WS_G    (WS_VT  + KTD*Nn*2)       // 2048 x 6144 bf16: G[(j*8+m)][(k,t,d)]
#define WS_H    (WS_G   + NTD*KTD*2)      // 256 x 2048 f32  : h_pre
#define WS_S    (WS_H   + Nn*NTD*4)       // 256 x 256 f32   : scores
#define WS_ATTB (WS_S   + Nn*Nn*4)        // 256 x 256 bf16  : softmax(att) bf16
#define WS_B    (WS_ATTB+ Nn*Nn*2)        // 256 x 6144 f32  : b = att @ V

// ============================================================
// Kernel 1: qkv projections. One block per batch b = n*8+k.
// Stages x[n] (12x32 f32) into LDS via the Tensor Data Mover.
// ============================================================
__global__ void qkv_kernel(const float* __restrict__ x,
                           const float* __restrict__ Wq, const float* __restrict__ Wk,
                           const float* __restrict__ Wv,
                           const float* __restrict__ bq, const float* __restrict__ bk,
                           const float* __restrict__ bv,
                           __bf16* __restrict__ Q, __bf16* __restrict__ Kk,
                           __bf16* __restrict__ VT) {
    __shared__ float xs[Tt * Ff];
    const int b   = blockIdx.x;
    const int n   = b >> 3;
    const int kb  = b & 7;
    const int tid = threadIdx.x;

#if HAVE_TDM
    if (tid < 32) {  // wave 0 issues one TDM descriptor for the 12x32 f32 tile
        unsigned long long ga = (unsigned long long)(uintptr_t)(x + (size_t)n * (Tt * Ff));
        unsigned int lds = (unsigned int)(size_t)&xs[0];
        u32x4 g0;
        g0[0] = 1u;                                           // count=1, user mode
        g0[1] = lds;                                          // lds_addr
        g0[2] = (unsigned int)(ga & 0xffffffffull);           // global_addr[31:0]
        g0[3] = (unsigned int)((ga >> 32) & 0x01ffffffull) | (2u << 30); // addr hi + type=2
        i32x8 g1;
        g1[0] = (2 << 16);            // data_size = 4 bytes
        g1[1] = (int)(32u << 16);     // tensor_dim0 = 32 (lo16 at bit48)
        g1[2] = (int)(12u << 16);     // tensor_dim1 = 12 (lo16 at bit80)
        g1[3] = (int)(32u << 16);     // tile_dim0   = 32 (at bit112)
        g1[4] = 12;                   // tile_dim1   = 12
        g1[5] = 32;                   // tensor_dim0_stride = 32
        g1[6] = 0;
        g1[7] = 0;
        i32x4 z4 = {};
#if defined(__clang_major__) && __clang_major__ >= 23
        i32x8 z8 = {};
        __builtin_amdgcn_tensor_load_to_lds(g0, g1, z4, z4, z8, 0);
#else
        __builtin_amdgcn_tensor_load_to_lds(g0, g1, z4, z4, 0);
#endif
    }
    __builtin_amdgcn_s_wait_tensorcnt(0);
    __syncthreads();
#else
    for (int idx = tid; idx < Tt * Ff; idx += 256) xs[idx] = x[(size_t)n * (Tt * Ff) + idx];
    __syncthreads();
#endif

    const float* wqb = Wq + (size_t)b * (Ff * Dd);
    const float* wkb = Wk + (size_t)b * (Ff * Dd);
    const float* wvb = Wv + (size_t)b * (Ff * Dd);

    for (int o = tid; o < Tt * Dd; o += 256) {
        const int t = o >> 6, d = o & 63;
        float aq = bq[(size_t)b * Dd + d];
        float ak = bk[(size_t)b * Dd + d];
        float av = bv[(size_t)b * Dd + d];
        #pragma unroll 8
        for (int f = 0; f < Ff; ++f) {
            const float xv = xs[t * Ff + f];
            aq = fmaf(xv, wqb[f * Dd + d], aq);
            ak = fmaf(xv, wkb[f * Dd + d], ak);
            av = fmaf(xv, wvb[f * Dd + d], av);
        }
        const int col = kb * (Tt * Dd) + t * Dd + d;           // (k,t,d) flat
        Q [(size_t)n * KTD + col] = (__bf16)aq;
        Kk[(size_t)n * KTD + col] = (__bf16)ak;
        VT[(size_t)col * Nn + n]  = (__bf16)av;                // transposed for GEMM2 B-operand
    }
}

// ============================================================
// Kernel 2: fold w1 into k:
//   G[(j*8+m)][(k,t,d)] = sum_s w1[m][k*144+t*12+s] * K[j][(k,s,d)]
// ============================================================
__global__ void g_kernel(const __bf16* __restrict__ Kk, const float* __restrict__ w1,
                         __bf16* __restrict__ G) {
    __shared__ float ks[Tt * Dd];      // 12 x 64
    __shared__ float w1s[TD * Tt * Tt];// 8 x 144
    const int j  = blockIdx.x >> 3;
    const int kb = blockIdx.x & 7;
    const int tid = threadIdx.x;

    for (int idx = tid; idx < Tt * Dd; idx += 256)
        ks[idx] = (float)Kk[(size_t)j * KTD + kb * (Tt * Dd) + idx];
    for (int idx = tid; idx < TD * Tt * Tt; idx += 256) {
        const int m = idx / (Tt * Tt), ts = idx % (Tt * Tt);
        w1s[idx] = w1[(size_t)m * (TD * Tt * Tt) + kb * (Tt * Tt) + ts];
    }
    __syncthreads();

    for (int o = tid; o < TD * Tt * Dd; o += 256) {
        const int m = o / (Tt * Dd);
        const int rem = o % (Tt * Dd);
        const int t = rem >> 6, d = rem & 63;
        float acc = 0.f;
        #pragma unroll
        for (int s = 0; s < Tt; ++s)
            acc = fmaf(w1s[m * (Tt * Tt) + t * Tt + s], ks[s * Dd + d], acc);
        G[((size_t)j * 8 + m) * KTD + kb * (Tt * Dd) + t * Dd + d] = (__bf16)acc;
    }
}

// ------------------------------------------------------------
// WMMA tile loaders per the CDNA5 16-bit A(16x32)/B(32x16) lane
// layouts. `o` is compile-time after unrolling, so loads lower to
// global_load_b128 with immediate IOFFSET against base VGPRs that
// are bumped once per 256-element macro step.
// ------------------------------------------------------------
__device__ __forceinline__ v16bf load_a_tile(const __bf16* __restrict__ p, int o, int koff) {
    v8bf lo = *(const v8bf*)(p + o + koff);
    v8bf hi = *(const v8bf*)(p + o + koff + 16);
    return __builtin_shufflevector(lo, hi, 0,1,2,3,4,5,6,7,8,9,10,11,12,13,14,15);
}
__device__ __forceinline__ v16bf load_b_tile(const __bf16* __restrict__ p, int o, int koff) {
    v8bf lo = *(const v8bf*)(p + o + koff);
    v8bf hi = *(const v8bf*)(p + o + koff + 16);
    return __builtin_shufflevector(lo, hi, 0,1,2,3,4,5,6,7,8,9,10,11,12,13,14,15);
}

#define WMMA_BF16(A, B, C) \
    __builtin_amdgcn_wmma_f32_16x16x32_bf16(false, (A), false, (B), (short)0, (C), false, false)

// NT bf16 GEMM core, one wave -> 16(M) x 64(N) tile, software-pipelined
// one k-step deep: step u issues the loads for step u+1 into fresh
// temporaries BEFORE consuming step u's tiles, so each WMMA finds ~10
// younger loads outstanding and s_wait_loadcnt becomes a partial drain
// instead of 0x0. The cur<-next copies carry distinct register sets
// across the unrolled iterations (no regalloc coalescing). Macro-step
// boundary resolved at compile time (u<7); tail is a clamped redundant
// reload (branch-free, EXEC stays all-ones around WMMA).
template<int KDIM>
__device__ __forceinline__ void gemm_nt_16x64(const __bf16* __restrict__ arow,
                                              const __bf16* const brow[4],
                                              int lane, v8f acc[4]) {
    const int koffA = (lane >> 4) * 8;    // lanes 0-15: K 0..7,16..23 ; 16-31: 8..15,24..31
    const int koffB = (lane >> 4) * 8;    // lanes 0-15: K 0..7,16..23 ; 16-31: 8..15,24..31

    const __bf16* ap  = arow;
    const __bf16* bp0 = brow[0];
    const __bf16* bp1 = brow[1];
    const __bf16* bp2 = brow[2];
    const __bf16* bp3 = brow[3];

    v16bf a_c  = load_a_tile(ap,  0, koffA);
    v16bf b0_c = load_b_tile(bp0, 0, koffB);
    v16bf b1_c = load_b_tile(bp1, 0, koffB);
    v16bf b2_c = load_b_tile(bp2, 0, koffB);
    v16bf b3_c = load_b_tile(bp3, 0, koffB);

    for (int kk0 = 0; kk0 < KDIM; kk0 += 256) {
        const int nb = (kk0 + 256 < KDIM) ? (kk0 + 256) : kk0;   // clamped tail
        const __bf16* apn  = arow    + nb;
        const __bf16* bpn0 = brow[0] + nb;
        const __bf16* bpn1 = brow[1] + nb;
        const __bf16* bpn2 = brow[2] + nb;
        const __bf16* bpn3 = brow[3] + nb;
        __builtin_prefetch(apn  + 256, 0, 3);
        __builtin_prefetch(bpn0 + 256, 0, 3);
        __builtin_prefetch(bpn1 + 256, 0, 3);
        __builtin_prefetch(bpn2 + 256, 0, 3);
        __builtin_prefetch(bpn3 + 256, 0, 3);

        #pragma unroll
        for (int u = 0; u < 8; ++u) {
            v16bf a_n, b0_n, b1_n, b2_n, b3_n;
            if (u < 7) {                     // compile-time: in-step immediate offsets
                const int on = (u + 1) * 32;
                a_n  = load_a_tile(ap,  on, koffA);
                b0_n = load_b_tile(bp0, on, koffB);
                b1_n = load_b_tile(bp1, on, koffB);
                b2_n = load_b_tile(bp2, on, koffB);
                b3_n = load_b_tile(bp3, on, koffB);
            } else {                         // first step of next macro block
                a_n  = load_a_tile(apn,  0, koffA);
                b0_n = load_b_tile(bpn0, 0, koffB);
                b1_n = load_b_tile(bpn1, 0, koffB);
                b2_n = load_b_tile(bpn2, 0, koffB);
                b3_n = load_b_tile(bpn3, 0, koffB);
            }
            acc[0] = WMMA_BF16(a_c, b0_c, acc[0]);
            acc[1] = WMMA_BF16(a_c, b1_c, acc[1]);
            acc[2] = WMMA_BF16(a_c, b2_c, acc[2]);
            acc[3] = WMMA_BF16(a_c, b3_c, acc[3]);
            a_c  = a_n;
            b0_c = b0_n; b1_c = b1_n; b2_c = b2_n; b3_c = b3_n;
        }
        ap = apn; bp0 = bpn0; bp1 = bpn1; bp2 = bpn2; bp3 = bpn3;
    }
}

// Store a 16x64 f32 tile per the 8-VGPR C/D layout.
__device__ __forceinline__ void store_c_16x64(float* __restrict__ C, int ldc,
                                              int m0, int n0, int lane, const v8f acc[4]) {
    const int r0 = m0 + (lane >> 4) * 8;
    #pragma unroll
    for (int t = 0; t < 4; ++t) {
        const int col = n0 + t * 16 + (lane & 15);
        #pragma unroll
        for (int r = 0; r < 8; ++r)
            C[(size_t)(r0 + r) * ldc + col] = acc[t][r];
    }
}

// ============================================================
// Kernel 3: h_pre = Q (256 x 6144) @ G^T (6144 x 2048), f32 acc.
// 16 mtiles x 32 ngroups = 512 waves.
// ============================================================
__global__ void gemm1_kernel(const __bf16* __restrict__ Q, const __bf16* __restrict__ G,
                             float* __restrict__ H) {
    const int lane = threadIdx.x & 31;
    const int wid  = blockIdx.x * 8 + (threadIdx.x >> 5);
    const int m0   = (wid >> 5) * 16;
    const int n0   = (wid & 31) * 64;

    const __bf16* arow = Q + (size_t)(m0 + (lane & 15)) * KTD;
    const __bf16* brow[4];
    #pragma unroll
    for (int t = 0; t < 4; ++t)
        brow[t] = G + (size_t)(n0 + t * 16 + (lane & 15)) * KTD;

    v8f acc[4] = {};
    gemm_nt_16x64<KTD>(arow, brow, lane, acc);
    store_c_16x64(H, NTD, m0, n0, lane, acc);
}

// ============================================================
// Kernel 4: s[i][j] = b2 + sum_m relu(h_pre[i][j*8+m]*norm + b1[m]) * w2[m]
// ============================================================
__global__ void score_kernel(const float* __restrict__ H, const float* __restrict__ b1,
                             const float* __restrict__ w2, const float* __restrict__ b2,
                             float* __restrict__ S) {
    const int i = blockIdx.x, j = threadIdx.x;
    const float norm = 0.125f;  // 1/sqrt(64)
    float acc = b2[0];
    #pragma unroll
    for (int m = 0; m < TD; ++m)
        acc = fmaf(fmaxf(H[(size_t)i * NTD + j * 8 + m] * norm + b1[m], 0.f), w2[m], acc);
    S[(size_t)i * Nn + j] = acc;
}

// ============================================================
// Kernel 5: row softmax; emit f32 att to d_out and bf16 copy for GEMM2.
// ============================================================
__global__ void softmax_kernel(const float* __restrict__ S, float* __restrict__ att_out,
                               __bf16* __restrict__ attb) {
    __shared__ float red[256];
    const int i = blockIdx.x, j = threadIdx.x;
    const float v = S[(size_t)i * Nn + j];
    red[j] = v; __syncthreads();
    for (int off = 128; off > 0; off >>= 1) {
        if (j < off) red[j] = fmaxf(red[j], red[j + off]);
        __syncthreads();
    }
    const float mx = red[0]; __syncthreads();
    const float e = __expf(v - mx);
    red[j] = e; __syncthreads();
    for (int off = 128; off > 0; off >>= 1) {
        if (j < off) red[j] += red[j + off];
        __syncthreads();
    }
    const float a = e / red[0];
    att_out[(size_t)i * Nn + j] = a;
    attb[(size_t)i * Nn + j]    = (__bf16)a;
}

// ============================================================
// Kernel 6: b = att (256x256) @ V (256x6144), f32 acc, via V^T rows.
// K=256 -> one pipelined macro step (32 WMMAs).
// 16 mtiles x 96 ngroups = 1536 waves.
// ============================================================
__global__ void gemm2_kernel(const __bf16* __restrict__ attb, const __bf16* __restrict__ VT,
                             float* __restrict__ B) {
    const int lane = threadIdx.x & 31;
    const int wid  = blockIdx.x * 8 + (threadIdx.x >> 5);
    const int m0   = (wid / 96) * 16;
    const int n0   = (wid % 96) * 64;

    const __bf16* arow = attb + (size_t)(m0 + (lane & 15)) * Nn;
    const __bf16* brow[4];
    #pragma unroll
    for (int t = 0; t < 4; ++t)
        brow[t] = VT + (size_t)(n0 + t * 16 + (lane & 15)) * Nn;

    v8f acc[4] = {};
    gemm_nt_16x64<Nn>(arow, brow, lane, acc);
    store_c_16x64(B, KTD, m0, n0, lane, acc);
}

// ============================================================
// Kernel 7: output MLP per (i,t) row.
// ============================================================
__global__ void out_kernel(const float* __restrict__ B, const float* __restrict__ w3,
                           const float* __restrict__ b3, const float* __restrict__ w4,
                           const float* __restrict__ b4, float* __restrict__ y) {
    __shared__ float in512[Dd * TD];
    __shared__ float y1[Dd];
    const int i = blockIdx.x / Tt;
    const int t = blockIdx.x % Tt;
    const int tid = threadIdx.x;  // 64 threads

    #pragma unroll
    for (int r = 0; r < 8; ++r) {
        const int idx = tid * 8 + r;          // idx = d*8 + k  (w3 input order)
        const int d = idx >> 3, k = idx & 7;
        in512[idx] = B[(size_t)i * KTD + k * (Tt * Dd) + t * Dd + d];
    }
    __syncthreads();

    {
        float acc = b3[tid];
        #pragma unroll 8
        for (int c = 0; c < Dd * TD; ++c)
            acc = fmaf(w3[(size_t)tid * (Dd * TD) + c], in512[c], acc);
        y1[tid] = fmaxf(acc, 0.f);
    }
    __syncthreads();

    if (tid < Ff) {
        float acc = b4[tid];
        #pragma unroll 8
        for (int d = 0; d < Dd; ++d)
            acc = fmaf(w4[(size_t)tid * Dd + d], y1[d], acc);
        y[((size_t)i * Tt + t) * Ff + tid] = acc;
    }
}

// ============================================================
extern "C" void kernel_launch(void* const* d_in, const int* in_sizes, int n_in,
                              void* d_out, int out_size, void* d_ws, size_t ws_size,
                              hipStream_t stream) {
    const float* x  = (const float*)d_in[0];
    const float* Wq = (const float*)d_in[1];
    const float* Wk = (const float*)d_in[2];
    const float* Wv = (const float*)d_in[3];
    const float* bq = (const float*)d_in[4];
    const float* bk = (const float*)d_in[5];
    const float* bv = (const float*)d_in[6];
    const float* w1 = (const float*)d_in[7];
    const float* b1 = (const float*)d_in[8];
    const float* w2 = (const float*)d_in[9];
    const float* b2 = (const float*)d_in[10];
    const float* w3 = (const float*)d_in[11];
    const float* b3 = (const float*)d_in[12];
    const float* w4 = (const float*)d_in[13];
    const float* b4 = (const float*)d_in[14];

    char* ws = (char*)d_ws;
    __bf16* Q    = (__bf16*)(ws + WS_Q);
    __bf16* Kf   = (__bf16*)(ws + WS_K);
    __bf16* VT   = (__bf16*)(ws + WS_VT);
    __bf16* G    = (__bf16*)(ws + WS_G);
    float*  H    = (float*) (ws + WS_H);
    float*  S    = (float*) (ws + WS_S);
    __bf16* attb = (__bf16*)(ws + WS_ATTB);
    float*  Bm   = (float*) (ws + WS_B);

    float* y_out   = (float*)d_out;                 // (256,12,32)
    float* att_out = (float*)d_out + Nn * Tt * Ff;  // (256,256)

    qkv_kernel<<<NTD, 256, 0, stream>>>(x, Wq, Wk, Wv, bq, bk, bv, Q, Kf, VT);
    g_kernel<<<NTD, 256, 0, stream>>>(Kf, w1, G);
    gemm1_kernel<<<64, 256, 0, stream>>>(Q, G, H);
    score_kernel<<<Nn, 256, 0, stream>>>(H, b1, w2, b2, S);
    softmax_kernel<<<Nn, 256, 0, stream>>>(S, att_out, attb);
    gemm2_kernel<<<192, 256, 0, stream>>>(attb, VT, Bm);
    out_kernel<<<Nn * Tt, 64, 0, stream>>>(Bm, w3, b3, w4, b4, y_out);
}
